// MultiHeadSelfAttention_71631464563424
// MI455X (gfx1250) — matrix operand order repaired and checked by reading, not compiled, not run on hardware
//
#include <hip/hip_runtime.h>

// ---------------------------------------------------------------------------
// MI455X (gfx1250) fused multi-head self-attention, bf16 WMMA + fp32 accum.
//   1) cast fp32 -> bf16 (x, Wqkv, Wout)
//   2) qkv = x @ Wqkv^T     (WMMA GEMM, async-to-LDS double buffered, bf16 out)
//   3) flash attention      (fused QK^T -> online softmax -> PV), bf16 y
//   4) out = y @ Wout^T     (WMMA GEMM, fp32 out -> d_out)
// ---------------------------------------------------------------------------

typedef __attribute__((ext_vector_type(16))) __bf16 v16bf;
typedef __attribute__((ext_vector_type(8)))  __bf16 v8bf;
typedef __attribute__((ext_vector_type(8)))  float  v8f;

#define D_MODEL 1024
#define E_QKV   3072
#define SEQ_T   2048
#define N_HEADS 16
#define D_HEAD  64
#define BT_ROWS 4096   // B*T

__device__ __forceinline__ v8f wmma_bf16(v16bf a, v16bf b, v8f c) {
  // D = A(16x32 bf16) * B(32x16 bf16) + C(16x16 f32)
  return __builtin_amdgcn_wmma_f32_16x16x32_bf16(
      /*neg_a=*/false, a, /*neg_b=*/false, b,
      /*c_mod=*/(short)0, c, /*reuse_a=*/false, /*reuse_b=*/false);
}

// CDNA5 async DMA: global memory -> LDS, 16 bytes per lane, tracked by
// ASYNCcnt. The LDS address VGPR takes the wave-relative LDS byte offset,
// which is addr[31:0] of the generic (flat) pointer to a __shared__ object.
__device__ __forceinline__ void async_load_b128(const void* gaddr, void* lds) {
  asm volatile("global_load_async_to_lds_b128 %0, %1, off"
               :: "v"((unsigned)(size_t)lds), "v"(gaddr)
               : "memory");
}
__device__ __forceinline__ void wait_async0() {
  asm volatile("s_wait_asynccnt 0x0" ::: "memory");
}

// A-matrix fragment (16x32, MxK). `row` points at this lane's M-row, K=0.
// ISA layout: lanes 0-15 hold K=0..7 (elems 0-7) and K=16..23 (elems 8-15);
// lanes 16-31 hold K=8..15 and K=24..31. Two 16-byte LDS loads.
__device__ __forceinline__ v16bf load_a_frag(const __bf16* row, int hi) {
  v8bf lo = *(const v8bf*)(row + 8 * hi);
  v8bf hh = *(const v8bf*)(row + 16 + 8 * hi);
  v16bf r;
#pragma unroll
  for (int i = 0; i < 8; ++i) { r[i] = lo[i]; r[8 + i] = hh[i]; }
  return r;
}

// B-matrix fragment (32x16, KxN). `col` points at this lane's N-column stored
// as a contiguous K-run. Lanes 0-15 hold K=0..7,16..23; lanes 16-31 hold K=8..15,24..31.
__device__ __forceinline__ v16bf load_b_frag(const __bf16* col, int hi) {
  v8bf lo = *(const v8bf*)(col + 8 * hi);
  v8bf hh = *(const v8bf*)(col + 16 + 8 * hi);
  v16bf r;
#pragma unroll
  for (int i = 0; i < 8; ++i) { r[i] = lo[i]; r[8 + i] = hh[i]; }
  return r;
}

// ---------------------------------------------------------------------------
// fp32 -> bf16 cast
// ---------------------------------------------------------------------------
__global__ __launch_bounds__(256) void cvt_f32_bf16(const float* __restrict__ in,
                                                    __bf16* __restrict__ out, int n) {
  int i = blockIdx.x * blockDim.x + threadIdx.x;
  if (i < n) out[i] = (__bf16)in[i];
}

// ---------------------------------------------------------------------------
// C[M,N] = A[M,K] * Bw[N,K]^T   (both bf16 row-major, fp32 accumulate)
// Block: 256 threads = 8 waves, tile 128(M) x 64(N), K-step 32.
// Double-buffered LDS filled by global_load_async_to_lds_b128; the DMA for
// step k+1 overlaps the 4-WMMA compute of step k. One barrier per K-step.
// ---------------------------------------------------------------------------
#define G_AST 40  // LDS row stride (bf16 elems): 80B, 16B-aligned, conflict pad
#define G_BST 40

template <bool OUT_BF16>
__global__ __launch_bounds__(256) void gemm_nt(const __bf16* __restrict__ A,
                                               const __bf16* __restrict__ Bw,
                                               void* __restrict__ Cout,
                                               int M, int N, int K) {
  __shared__ __align__(16) __bf16 As[2][128 * G_AST];
  __shared__ __align__(16) __bf16 Bs[2][64 * G_BST];

  const int n0   = blockIdx.x * 64;
  const int m0   = blockIdx.y * 128;
  const int tid  = threadIdx.x;
  const int wave = tid >> 5, lane = tid & 31;
  const int hi   = lane >> 4, lo16 = lane & 15;
  const int wm   = wave & 3, wn = wave >> 2;

  // per-thread chunk mapping (8 bf16 = 16B per chunk)
  const int ar0 = tid >> 2, aq = (tid & 3) * 8;          // A chunks: tid, tid+256
  const int br0 = tid >> 2, bq = (tid & 3) * 8;          // B chunk: tid

  auto issue_tile = [&](int k0, int buf) {
#pragma unroll
    for (int i = 0; i < 2; ++i) {
      int row = ar0 + i * 64;
      async_load_b128(A + (size_t)(m0 + row) * K + k0 + aq,
                      &As[buf][row * G_AST + aq]);
    }
    async_load_b128(Bw + (size_t)(n0 + br0) * K + k0 + bq,
                    &Bs[buf][br0 * G_BST + bq]);
  };

  v8f acc[2][2] = {};
  issue_tile(0, 0);
  int cur = 0;

  for (int k0 = 0; k0 < K; k0 += 32) {
    wait_async0();       // this wave's DMA chunks landed in LDS
    __syncthreads();     // every wave's chunks landed; prev readers done
    if (k0 + 32 < K) issue_tile(k0 + 32, cur ^ 1);

    v16bf af[2], bf[2];
#pragma unroll
    for (int i = 0; i < 2; ++i)
      af[i] = load_a_frag(&As[cur][(wm * 32 + i * 16 + lo16) * G_AST], hi);
#pragma unroll
    for (int j = 0; j < 2; ++j)
      bf[j] = load_b_frag(&Bs[cur][(wn * 32 + j * 16 + lo16) * G_BST], hi);

#pragma unroll
    for (int i = 0; i < 2; ++i)
#pragma unroll
      for (int j = 0; j < 2; ++j)
        acc[i][j] = wmma_bf16(af[i], bf[j], acc[i][j]);

    cur ^= 1;
  }

  // C/D layout: lane: col = lo16, vgpr r: row = r + 8*hi
#pragma unroll
  for (int i = 0; i < 2; ++i)
#pragma unroll
    for (int j = 0; j < 2; ++j) {
      int col = n0 + wn * 32 + j * 16 + lo16;
#pragma unroll
      for (int r = 0; r < 8; ++r) {
        int row = m0 + wm * 32 + i * 16 + r + 8 * hi;
        float v = acc[i][j][r];
        if (OUT_BF16)
          ((__bf16*)Cout)[(size_t)row * N + col] = (__bf16)v;
        else
          ((float*)Cout)[(size_t)row * N + col] = v;
      }
    }
}

// ---------------------------------------------------------------------------
// Flash attention: one block per (b, h, 64-row q tile). 128 threads = 4 waves;
// wave w owns q rows [w*16, w*16+16). Online softmax, causal mask on the
// diagonal tile only. Q/K tiles DMA'd row-major into LDS (async-to-LDS);
// V transposed into LDS so the PV B-fragment is a contiguous row; P
// round-trips through wave-private LDS rows to re-enter the WMMA A-layout.
// ---------------------------------------------------------------------------
#define F_ST 72  // LDS row stride (bf16): 144B, 16B-aligned

__global__ __launch_bounds__(128) void flash_attn(const __bf16* __restrict__ qkv,
                                                  __bf16* __restrict__ y) {
  __shared__ __align__(16) __bf16 Qs[64 * F_ST];
  __shared__ __align__(16) __bf16 Ks[64 * F_ST];
  __shared__ __align__(16) __bf16 Vt[64 * F_ST];  // Vt[dh][key]
  __shared__ __align__(16) __bf16 Ps[64 * F_ST];

  const int qt   = blockIdx.x;  // 0..31
  const int h    = blockIdx.y;  // 0..15
  const int b    = blockIdx.z;  // 0..1
  const int tid  = threadIdx.x;
  const int wave = tid >> 5, lane = tid & 31;
  const int hi   = lane >> 4, lo16 = lane & 15;
  const int q0   = qt * 64;
  const size_t rb = (size_t)b * SEQ_T;
  const int qcol = h * D_HEAD;
  const int kcol = D_MODEL + h * D_HEAD;
  const int vcol = 2 * D_MODEL + h * D_HEAD;
  const float scale = 0.125f;  // 1/sqrt(64)

  // Q tile 64x64: 512 16B-chunks, 4 async DMAs per thread
#pragma unroll
  for (int i = 0; i < 4; ++i) {
    int c = tid + i * 128;
    int row = c >> 3, q = (c & 7) * 8;
    async_load_b128(qkv + (rb + q0 + row) * E_QKV + qcol + q,
                    Qs + row * F_ST + q);
  }

  v8f o[4] = {};
  float mrow[8], lrow[8];
#pragma unroll
  for (int r = 0; r < 8; ++r) { mrow[r] = -1e30f; lrow[r] = 0.f; }

  for (int j = 0; j <= qt; ++j) {
    const int kb = j * 64;
    // K tile row-major: async DMA
#pragma unroll
    for (int i = 0; i < 4; ++i) {
      int c = tid + i * 128;
      int row = c >> 3, q = (c & 7) * 8;
      async_load_b128(qkv + (rb + kb + row) * E_QKV + kcol + q,
                      Ks + row * F_ST + q);
    }
    // V tile transposed: Vt[dh][key] (register path, scalar LDS stores)
#pragma unroll
    for (int i = 0; i < 4; ++i) {
      int c = tid + i * 128;
      int row = c >> 3, q = c & 7;  // key = row, dh = q*8 .. q*8+7
      uint4 d = *(const uint4*)(qkv + (rb + kb + row) * E_QKV + vcol + q * 8);
      const __bf16* e = (const __bf16*)&d;
#pragma unroll
      for (int t = 0; t < 8; ++t) Vt[(q * 8 + t) * F_ST + row] = e[t];
    }
    wait_async0();     // Q (first iter) + K DMAs landed
    __syncthreads();   // all waves' tiles visible; prev-iter readers done

    // S = Q * K^T   (16x64 per wave, 4 key-subtiles x 2 k-steps)
    v8f s[4];
#pragma unroll
    for (int n = 0; n < 4; ++n) {
      v8f a = {};
#pragma unroll
      for (int ss = 0; ss < 2; ++ss) {
        v16bf qa = load_a_frag(Qs + (wave * 16 + lo16) * F_ST + ss * 32, hi);
        v16bf kb2 = load_b_frag(Ks + (n * 16 + lo16) * F_ST + ss * 32, hi);
        a = wmma_bf16(qa, kb2, a);
      }
      s[n] = a;
    }

    // scale + causal mask (diagonal tile only)
    const bool diag = (j == qt);
#pragma unroll
    for (int n = 0; n < 4; ++n)
#pragma unroll
      for (int r = 0; r < 8; ++r) {
        float v = s[n][r] * scale;
        if (diag) {
          int qrow = q0 + wave * 16 + r + 8 * hi;
          int kidx = kb + n * 16 + lo16;
          v = (kidx > qrow) ? -1e30f : v;
        }
        s[n][r] = v;
      }

    // online softmax; row M=r+8*hi lives on the 16 lanes of half `hi`,
    // so xor masks 1,2,4,8 reduce exactly within the row.
    float resc[8];
#pragma unroll
    for (int r = 0; r < 8; ++r) {
      float mx = fmaxf(fmaxf(s[0][r], s[1][r]), fmaxf(s[2][r], s[3][r]));
#pragma unroll
      for (int m = 8; m >= 1; m >>= 1) mx = fmaxf(mx, __shfl_xor(mx, m, 32));
      float mnew = fmaxf(mrow[r], mx);
      resc[r] = __expf(mrow[r] - mnew);
      mrow[r] = mnew;
      float sum = 0.f;
#pragma unroll
      for (int n = 0; n < 4; ++n) {
        float p = __expf(s[n][r] - mnew);
        s[n][r] = p;
        sum += p;
      }
#pragma unroll
      for (int m = 8; m >= 1; m >>= 1) sum += __shfl_xor(sum, m, 32);
      lrow[r] = lrow[r] * resc[r] + sum;
    }

    // P -> wave-private LDS rows (re-enter A-layout); rescale O
#pragma unroll
    for (int n = 0; n < 4; ++n)
#pragma unroll
      for (int r = 0; r < 8; ++r)
        Ps[(wave * 16 + r + 8 * hi) * F_ST + n * 16 + lo16] = (__bf16)s[n][r];
#pragma unroll
    for (int n = 0; n < 4; ++n)
#pragma unroll
      for (int r = 0; r < 8; ++r) o[n][r] *= resc[r];

    // O += P * V   (4 dh-subtiles x 2 k-steps over keys)
#pragma unroll
    for (int n = 0; n < 4; ++n)
#pragma unroll
      for (int ss = 0; ss < 2; ++ss) {
        v16bf pa = load_a_frag(Ps + (wave * 16 + lo16) * F_ST + ss * 32, hi);
        v16bf vb = load_b_frag(Vt + (n * 16 + lo16) * F_ST + ss * 32, hi);
        o[n] = wmma_bf16(pa, vb, o[n]);
      }

    __syncthreads();  // Ks/Vt consumed, safe to reload next j
  }

  // normalize and store y[b, t, h*64 + dh] as bf16
#pragma unroll
  for (int n = 0; n < 4; ++n)
#pragma unroll
    for (int r = 0; r < 8; ++r) {
      int trow = q0 + wave * 16 + r + 8 * hi;
      float v = o[n][r] / lrow[r];
      y[(rb + trow) * D_MODEL + h * D_HEAD + n * 16 + lo16] = (__bf16)v;
    }
}

// ---------------------------------------------------------------------------
// Host launcher
// ---------------------------------------------------------------------------
extern "C" void kernel_launch(void* const* d_in, const int* in_sizes, int n_in,
                              void* d_out, int out_size, void* d_ws, size_t ws_size,
                              hipStream_t stream) {
  (void)in_sizes; (void)n_in; (void)out_size; (void)ws_size;
  const float* x    = (const float*)d_in[0];
  const float* Wqkv = (const float*)d_in[1];
  const float* Wout = (const float*)d_in[2];
  float* out = (float*)d_out;

  char* ws = (char*)d_ws;
  size_t off = 0;
  auto carve = [&](size_t elems) -> __bf16* {
    __bf16* p = (__bf16*)(ws + off);
    off += ((elems * sizeof(__bf16)) + 255) & ~(size_t)255;
    return p;
  };
  __bf16* Xb    = carve((size_t)BT_ROWS * D_MODEL);   // 8 MB
  __bf16* Wqkvb = carve((size_t)E_QKV * D_MODEL);     // 6 MB
  __bf16* Woutb = carve((size_t)D_MODEL * D_MODEL);   // 2 MB
  __bf16* qkvb  = carve((size_t)BT_ROWS * E_QKV);     // 24 MB
  __bf16* yb    = carve((size_t)BT_ROWS * D_MODEL);   // 8 MB

  const int nx = BT_ROWS * D_MODEL;      // 4,194,304
  const int nq = E_QKV * D_MODEL;        // 3,145,728
  const int no = D_MODEL * D_MODEL;      // 1,048,576
  cvt_f32_bf16<<<(nx + 255) / 256, 256, 0, stream>>>(x, Xb, nx);
  cvt_f32_bf16<<<(nq + 255) / 256, 256, 0, stream>>>(Wqkv, Wqkvb, nq);
  cvt_f32_bf16<<<(no + 255) / 256, 256, 0, stream>>>(Wout, Woutb, no);

  // qkv = x @ Wqkv^T  -> bf16 [4096, 3072]
  gemm_nt<true><<<dim3(E_QKV / 64, BT_ROWS / 128), 256, 0, stream>>>(
      Xb, Wqkvb, qkvb, BT_ROWS, E_QKV, D_MODEL);

  // fused causal attention -> bf16 y [4096, 1024]
  flash_attn<<<dim3(SEQ_T / 64, N_HEADS, 2), 128, 0, stream>>>(qkvb, yb);

  // out = y @ Wout^T -> fp32 d_out
  gemm_nt<false><<<dim3(D_MODEL / 64, BT_ROWS / 128), 256, 0, stream>>>(
      yb, Woutb, out, BT_ROWS, D_MODEL, D_MODEL);
}
